// SpectralMambaBackbone_35948876268410
// MI455X (gfx1250) — matrix operand order repaired and checked by reading, not compiled, not run on hardware
//
#include <hip/hip_runtime.h>
#include <hip/hip_bf16.h>

// ---------------------------------------------------------------------------
// SpectralMambaBackbone for MI455X (gfx1250, wave32, WMMA + TDM).
//
// Bandwidth-bound workload (~177 MB activations vs 10 GFLOP) -> keep the
// workspace L2-resident, run all GEMMs through v_wmma_f32_16x16x32_f16
// (f16 fragments, f32 accumulate), avoid flip copies by direction-
// parameterized conv/scan, and stage the scan's dt/B/C slab into LDS with a
// single Tensor-Data-Mover descriptor (tensor_load_to_lds + s_wait_tensorcnt)
// so the 100-step recurrence runs barrier-free out of LDS.
// ---------------------------------------------------------------------------

#define NSEQ   1024        // B*H*W spatial positions -> independent sequences
#define LSEQ   100         // spectral channels = sequence length
#define DM     32          // d_model
#define DI     64          // d_inner
#define DSTATE 16          // d_state
#define DTR    2           // dt_rank
#define DBLN   (DTR + 2*DSTATE)   // 34 (x_proj output cols)
#define DBLP   48                 // padded row stride for the 34-col buffer
#define MROWS  (NSEQ*LSEQ)        // 102400 GEMM rows
#define SCAN_SLAB (4 * LSEQ * DBLP)   // 19200 f32 = 76.8 KB per workgroup

typedef _Float16 v16h __attribute__((ext_vector_type(16)));
typedef float    v8f  __attribute__((ext_vector_type(8)));
typedef unsigned int v4u __attribute__((ext_vector_type(4)));
typedef int          v4i __attribute__((ext_vector_type(4)));
typedef int          v8i __attribute__((ext_vector_type(8)));

#if defined(__has_builtin)
#if __has_builtin(__builtin_amdgcn_tensor_load_to_lds)
#define HAVE_TDM 1
#else
#define HAVE_TDM 0
#endif
#else
#define HAVE_TDM 0
#endif

// ---------------------------------------------------------------------------
// Generic WMMA GEMM:  C[M,N] (+)= A[M,K] @ B[K,N] (+ bias[n])
// f32 in memory, f16 fragments, f32 accumulation via v_wmma_f32_16x16x32_f16.
// 256 threads = 8 waves; block tile 64(M) x 32(N); wave tile 16x16.
// K must be 32 or 64 (all GEMMs here satisfy this; no K remainder).
// ---------------------------------------------------------------------------
#define KMAX 64
__global__ void __launch_bounds__(256)
gemm_wmma_kernel(const float* __restrict__ A, int lda,
                 const float* __restrict__ B, int ldb,
                 float* __restrict__ C, int ldc,
                 int M, int N, int K,
                 const float* __restrict__ bias, int accum)
{
    __shared__ _Float16 sA [64][KMAX + 8];   // A tile, row-major [m][k]
    __shared__ _Float16 sBt[32][KMAX + 8];   // B tile transposed [n][k]

    const int tid    = threadIdx.x;
    const int mBlock = blockIdx.y * 64;
    const int nBlock = blockIdx.x * 32;
    const int kshift = (K >= 64) ? 6 : 5;    // K in {32, 64}
    const int kmask  = K - 1;

    // Hint the near-term global panels into cache (global_prefetch_b8).
    __builtin_prefetch(A + (size_t)mBlock * lda, 0, 3);
    __builtin_prefetch(B, 0, 3);

    // Cooperative load of A (64 x K) and B^T (32 x K) into LDS as f16.
    for (int i = tid; i < 64 * K; i += 256) {
        int r = i >> kshift, k = i & kmask;
        int gm = mBlock + r;
        sA[r][k] = (_Float16)((gm < M) ? A[(size_t)gm * lda + k] : 0.0f);
    }
    for (int i = tid; i < 32 * K; i += 256) {
        int n = i >> kshift, k = i & kmask;
        int gn = nBlock + n;
        sBt[n][k] = (_Float16)((gn < N) ? B[(size_t)k * ldb + gn] : 0.0f);
    }
    __syncthreads();

    const int wave = tid >> 5;          // 0..7
    const int lane = tid & 31;
    const int mw   = (wave >> 1) * 16;  // 0,16,32,48
    const int nw   = (wave & 1) * 16;   // 0,16
    const int row  = lane & 15;         // M for A-frag, N for B-frag, N for C
    const int hiHalf = lane >> 4;       // 0 for lanes 0-15, 1 for 16-31
    const int klA = hiHalf * 8;         // unused; A-frag loads use klB (contiguous k)
    const int klB = hiHalf * 16;        // A/B-frag K base offset (contiguous k = 16h + s)

    v8f acc = {};
    for (int k0 = 0; k0 < K; k0 += 32) {
        v16h a, b;
#pragma unroll
        for (int i = 0; i < 8; ++i) {
            a[i]     = sA[mw + row][k0 + klB + i];        // K = klB + 0..7
            a[i + 8] = sA[mw + row][k0 + klB + 8 + i];    // K = klB + 8..15
        }
#pragma unroll
        for (int i = 0; i < 16; ++i) {
            b[i] = sBt[nw + row][k0 + klB + i];           // K = klB + 0..15
        }
        acc = __builtin_amdgcn_wmma_f32_16x16x32_f16(
                  /*neg_a=*/false, a, /*neg_b=*/false, b,
                  /*c_mod=*/(short)0, acc,
                  /*reuse_a=*/false, /*reuse_b=*/false);
    }

    // C/D layout: VGPR r -> M = r (+8 for lanes >= 16); N = lane & 15.
    const int cn = nBlock + nw + row;
    if (cn < N) {
        const int mbase = mBlock + mw + hiHalf * 8;
#pragma unroll
        for (int r = 0; r < 8; ++r) {
            int cm = mbase + r;
            if (cm < M) {
                float v = acc[r];
                if (bias)  v += bias[cn];
                float* p = &C[(size_t)cm * ldc + cn];
                if (accum) v += *p;
                *p = v;
            }
        }
    }
}

// ---------------------------------------------------------------------------
// seq[n,l,d] = x_spec[l, n] * in_w[d] + in_b[d]      (n = h*32+w)
// ---------------------------------------------------------------------------
__global__ void embed_kernel(const float* __restrict__ xs,
                             const float* __restrict__ in_w,
                             const float* __restrict__ in_b,
                             float* __restrict__ seq)
{
    int idx = blockIdx.x * 256 + threadIdx.x;
    if (idx >= NSEQ * LSEQ * DM) return;
    int d = idx & (DM - 1);
    int r = idx >> 5;              // n*LSEQ + l
    int l = r % LSEQ;
    int n = r / LSEQ;
    seq[idx] = xs[(size_t)l * NSEQ + n] * in_w[d] + in_b[d];
}

// ---------------------------------------------------------------------------
// LayerNorm over the 32-wide feature dim, one row per thread.
// ---------------------------------------------------------------------------
__global__ void layernorm_kernel(const float* __restrict__ x,
                                 const float* __restrict__ g,
                                 const float* __restrict__ b,
                                 float* __restrict__ o)
{
    int row = blockIdx.x * 256 + threadIdx.x;
    if (row >= MROWS) return;
    const float* xr = x + (size_t)row * DM;
    float m = 0.0f;
#pragma unroll
    for (int d = 0; d < DM; ++d) m += xr[d];
    m *= (1.0f / DM);
    float v = 0.0f;
#pragma unroll
    for (int d = 0; d < DM; ++d) { float t = xr[d] - m; v += t * t; }
    v *= (1.0f / DM);
    float inv = rsqrtf(v + 1e-5f);
    float* orow = o + (size_t)row * DM;
#pragma unroll
    for (int d = 0; d < DM; ++d) orow[d] = (xr[d] - m) * inv * g[d] + b[d];
}

// ---------------------------------------------------------------------------
// Depthwise causal conv (D_CONV=4) + SiLU. dir=1 runs the time-reversed
// (anti-causal) variant so the bwd mamba never materializes flip().
// ---------------------------------------------------------------------------
__global__ void conv_silu_kernel(const float* __restrict__ xi,
                                 const float* __restrict__ w,
                                 const float* __restrict__ cb,
                                 float* __restrict__ xc, int dir)
{
    int idx = blockIdx.x * 256 + threadIdx.x;
    if (idx >= NSEQ * LSEQ * DI) return;
    int d = idx & (DI - 1);
    int r = idx >> 6;
    int l = r % LSEQ;
    int n = r / LSEQ;
    float acc = cb[d];
#pragma unroll
    for (int k = 0; k < 4; ++k) {
        int off = 3 - k;
        int sl = dir ? (l + off) : (l - off);
        if (sl >= 0 && sl < LSEQ)
            acc += w[d * 4 + k] * xi[(((size_t)n * LSEQ + sl) << 6) + d];
    }
    xc[idx] = acc / (1.0f + __expf(-acc));   // SiLU
}

// ---------------------------------------------------------------------------
// delta = softplus(dt @ dt_w + dt_b), dt = first 2 cols of the padded dbl buf.
// ---------------------------------------------------------------------------
__global__ void delta_kernel(const float* __restrict__ dbl,
                             const float* __restrict__ dtw,
                             const float* __restrict__ dtb,
                             float* __restrict__ delta)
{
    int idx = blockIdx.x * 256 + threadIdx.x;
    if (idx >= NSEQ * LSEQ * DI) return;
    int d   = idx & (DI - 1);
    int row = idx >> 6;
    float v = dbl[(size_t)row * DBLP + 0] * dtw[d]
            + dbl[(size_t)row * DBLP + 1] * dtw[DI + d]
            + dtb[d];
    delta[idx] = (v > 20.0f) ? v : log1pf(__expf(v));
}

// ---------------------------------------------------------------------------
// Selective scan: block = 4 sequences x 64 channels; 16-state h in registers.
// The whole [4 x LSEQ x DBLP] dt/B/C slab for this block is contiguous in
// global memory, so a single TDM descriptor DMAs it into LDS up front
// (tensor_load_to_lds, TENSORcnt), after which the 100 sequential steps run
// barrier-free against LDS. dir=1 iterates time in reverse, which composes
// the flip/unflip of the bidirectional branch exactly.
// ---------------------------------------------------------------------------
__global__ void __launch_bounds__(256)
scan_kernel(const float* __restrict__ delta, const float* __restrict__ dbl,
            const float* __restrict__ u,     const float* __restrict__ A_log,
            const float* __restrict__ Dv,    float* __restrict__ y, int dir)
{
    __shared__ float sDBL[SCAN_SLAB];   // 76.8 KB -> 4 workgroups per WGP
    const int q = threadIdx.x >> 6;          // sequence slot in block
    const int d = threadIdx.x & (DI - 1);    // channel
    const int n = blockIdx.x * 4 + q;
    const float* gsrc = dbl + (size_t)blockIdx.x * SCAN_SLAB;

#if HAVE_TDM
    if (threadIdx.x < 32) {   // wave 0 issues the DMA (EXEC all-ones)
        unsigned int lds_off = (unsigned int)(unsigned long long)&sDBL[0];
        unsigned long long ga = (unsigned long long)gsrc;
        // D# group 0: count=1, lds_addr, 57-bit global_addr, type=2 (image).
        v4u g0 = { 1u,
                   lds_off,
                   (unsigned int)ga,
                   (unsigned int)((ga >> 32) & 0x01FFFFFFu) | 0x80000000u };
        // D# group 1: data_size=2 (4B), tensor_dim0 = tile_dim0 = SCAN_SLAB,
        // tensor_dim1 = tile_dim1 = 1, stride0 = SCAN_SLAB (1-D contiguous).
        v8i g1 = { (int)(2u << 16),                               // data_size
                   (int)((SCAN_SLAB & 0xFFFFu) << 16),            // dim0 lo16
                   (int)(((SCAN_SLAB >> 16) & 0xFFFFu) | (1u << 16)), // dim0 hi / dim1 lo
                   (int)((SCAN_SLAB & 0xFFFFu) << 16),            // tile_dim0
                   (int)1,                                        // tile_dim1
                   (int)SCAN_SLAB,                                // stride0 lo32
                   0, 0 };
        v4i z4 = { 0, 0, 0, 0 };
#if defined(__clang_major__) && (__clang_major__ >= 23)
        v8i z8 = { 0, 0, 0, 0, 0, 0, 0, 0 };
        __builtin_amdgcn_tensor_load_to_lds(g0, g1, z4, z4, z8, 0);
#else
        __builtin_amdgcn_tensor_load_to_lds(g0, g1, z4, z4, 0);
#endif
        __builtin_amdgcn_s_wait_tensorcnt(0);
    }
    __syncthreads();
#else
    for (int i = threadIdx.x; i < SCAN_SLAB; i += 256) sDBL[i] = gsrc[i];
    __syncthreads();
#endif

    float a[DSTATE], h[DSTATE];
#pragma unroll
    for (int s = 0; s < DSTATE; ++s) {
        a[s] = -__expf(A_log[d * DSTATE + s]);   // A = -exp(A_log)
        h[s] = 0.0f;
    }
    const float Dd = Dv[d];

    for (int t = 0; t < LSEQ; ++t) {
        int l = dir ? (LSEQ - 1 - t) : t;
        size_t row = (size_t)n * LSEQ + l;
        const float* rp = &sDBL[(q * LSEQ + l) * DBLP];   // [dt0 dt1 | B | C]

        float dt = delta[(row << 6) + d];
        float uu = u[(row << 6) + d];
        float du = dt * uu;
        float acc = 0.0f;
#pragma unroll
        for (int s = 0; s < DSTATE; ++s) {
            h[s] = h[s] * __expf(dt * a[s]) + du * rp[DTR + s];
            acc += h[s] * rp[DTR + DSTATE + s];
        }
        y[(row << 6) + d] = acc + uu * Dd;
    }
}

// y *= silu(z)
__global__ void gate_kernel(float* __restrict__ y, const float* __restrict__ z)
{
    int idx = blockIdx.x * 256 + threadIdx.x;
    if (idx >= NSEQ * LSEQ * DI) return;
    float zz = z[idx];
    y[idx] *= zz / (1.0f + __expf(-zz));
}

// ---------------------------------------------------------------------------
// Softmax attention pooling over the spectral axis, one block per sequence.
// ---------------------------------------------------------------------------
__global__ void __launch_bounds__(128)
attnpool_kernel(const float* __restrict__ seq, const float* __restrict__ aw,
                const float* __restrict__ ab, float* __restrict__ pooled)
{
    __shared__ float logits[LSEQ];
    __shared__ float invsum;
    const int n = blockIdx.x;
    const int t = threadIdx.x;

    if (t < LSEQ) {
        const float* sr = seq + ((size_t)n * LSEQ + t) * DM;
        float v = ab[0];
#pragma unroll
        for (int d = 0; d < DM; ++d) v += sr[d] * aw[d];
        logits[t] = v;
    }
    __syncthreads();
    if (t == 0) {
        float mx = -1e30f;
        for (int l = 0; l < LSEQ; ++l) mx = fmaxf(mx, logits[l]);
        float sm = 0.0f;
        for (int l = 0; l < LSEQ; ++l) { logits[l] = __expf(logits[l] - mx); sm += logits[l]; }
        invsum = 1.0f / sm;
    }
    __syncthreads();
    if (t < DM) {
        float acc = 0.0f;
        for (int l = 0; l < LSEQ; ++l)
            acc += logits[l] * seq[((size_t)n * LSEQ + l) * DM + t];
        pooled[(size_t)n * DM + t] = acc * invsum;
    }
}

// out[1,64,32,32]: d_out[f*1024 + n] = outTile[n*64 + f]
__global__ void writeout_kernel(const float* __restrict__ ot, float* __restrict__ out)
{
    int idx = blockIdx.x * 256 + threadIdx.x;
    if (idx >= NSEQ * 64) return;
    int f = idx & 63, n = idx >> 6;
    out[(size_t)f * NSEQ + n] = ot[((size_t)n << 6) + f];
}

// ---------------------------------------------------------------------------
// Host-side orchestration.
// ---------------------------------------------------------------------------
extern "C" void kernel_launch(void* const* d_in, const int* in_sizes, int n_in,
                              void* d_out, int out_size, void* d_ws, size_t ws_size,
                              hipStream_t stream) {
    (void)in_sizes; (void)n_in; (void)out_size; (void)ws_size;

    auto F = [&](int i) { return (const float*)d_in[i]; };

    // jax tree_flatten order (dict keys sorted):
    // 0 attn_b, 1 attn_w, then blocks[0..1] each = {bwd(9), fwd(9), ln_b, ln_g},
    // then in_b, in_w, pool_b, pool_w, x_spec.
    // mamba leaves sorted: A_log, D, conv_b, conv_w, dt_b, dt_w, in_proj, out_proj, x_proj
    struct MP { const float *A_log,*D,*conv_b,*conv_w,*dt_b,*dt_w,*in_proj,*out_proj,*x_proj; };
    auto loadMP = [&](int base) {
        MP p; p.A_log = F(base); p.D = F(base+1); p.conv_b = F(base+2);
        p.conv_w = F(base+3); p.dt_b = F(base+4); p.dt_w = F(base+5);
        p.in_proj = F(base+6); p.out_proj = F(base+7); p.x_proj = F(base+8);
        return p;
    };
    const float* attn_b = F(0);
    const float* attn_w = F(1);
    MP bwd[2], fwd[2];
    const float *lnb[2], *lng[2];
    for (int b = 0; b < 2; ++b) {
        int base = 2 + b * 20;
        bwd[b] = loadMP(base);
        fwd[b] = loadMP(base + 9);
        lnb[b] = F(base + 18);
        lng[b] = F(base + 19);
    }
    const float* in_b   = F(42);
    const float* in_w   = F(43);
    const float* pool_b = F(44);
    const float* pool_w = F(45);
    const float* x_spec = F(46);

    // Workspace carve (floats). Total ~177 MB -> stays resident in 192 MB L2.
    float* w = (float*)d_ws;
    size_t off = 0;
    auto carve = [&](size_t n) { float* p = w + off; off += n; return p; };
    float* SEQ    = carve((size_t)MROWS * DM);    // running residual stream
    float* XN     = carve((size_t)MROWS * DM);    // layernorm output
    float* XI     = carve((size_t)MROWS * DI);    // in_proj x-half
    float* Z      = carve((size_t)MROWS * DI);    // in_proj z-half (gate)
    float* XC     = carve((size_t)MROWS * DI);    // conv+silu output u
    float* DBL    = carve((size_t)MROWS * DBLP);  // [dt|B|C] padded to 48
    float* DELTA  = carve((size_t)MROWS * DI);
    float* Y      = carve((size_t)MROWS * DI);
    float* POOLED = carve((size_t)NSEQ * DM);
    float* OUTT   = carve((size_t)NSEQ * 64);

    const int EW_DM = (MROWS * DM) / 256;   // 12800
    const int EW_DI = (MROWS * DI) / 256;   // 25600
    const int ROWS  = MROWS / 256;          // 400

    auto gemm = [&](const float* A, int lda, const float* B, int ldb,
                    float* C, int ldc, int M, int N, int K,
                    const float* bias, int accum) {
        dim3 g((N + 31) / 32, (M + 63) / 64);
        gemm_wmma_kernel<<<g, 256, 0, stream>>>(A, lda, B, ldb, C, ldc,
                                                M, N, K, bias, accum);
    };

    auto mamba = [&](const MP& p, int dir) {
        // xz = xn @ in_proj, split into x-half and z-half (two N=64 GEMMs)
        gemm(XN, DM, p.in_proj,      2 * DI, XI, DI, MROWS, DI, DM, nullptr, 0);
        gemm(XN, DM, p.in_proj + DI, 2 * DI, Z,  DI, MROWS, DI, DM, nullptr, 0);
        conv_silu_kernel<<<EW_DI, 256, 0, stream>>>(XI, p.conv_w, p.conv_b, XC, dir);
        gemm(XC, DI, p.x_proj, DBLN, DBL, DBLP, MROWS, DBLN, DI, nullptr, 0);
        delta_kernel<<<EW_DI, 256, 0, stream>>>(DBL, p.dt_w, p.dt_b, DELTA);
        scan_kernel<<<NSEQ / 4, 256, 0, stream>>>(DELTA, DBL, XC, p.A_log, p.D, Y, dir);
        gate_kernel<<<EW_DI, 256, 0, stream>>>(Y, Z);
        // residual: SEQ += (y * silu(z)) @ out_proj   (accumulating GEMM)
        gemm(Y, DI, p.out_proj, DM, SEQ, DM, MROWS, DM, DI, nullptr, 1);
    };

    embed_kernel<<<EW_DM, 256, 0, stream>>>(x_spec, in_w, in_b, SEQ);
    for (int b = 0; b < 2; ++b) {
        layernorm_kernel<<<ROWS, 256, 0, stream>>>(SEQ, lng[b], lnb[b], XN);
        mamba(fwd[b], /*dir=*/0);
        mamba(bwd[b], /*dir=*/1);
    }
    attnpool_kernel<<<NSEQ, 128, 0, stream>>>(SEQ, attn_w, attn_b, POOLED);
    gemm(POOLED, DM, pool_w, 64, OUTT, 64, NSEQ, 64, DM, pool_b, 0);
    writeout_kernel<<<(NSEQ * 64) / 256, 256, 0, stream>>>(OUTT, (float*)d_out);
}
